// MultiHeadAttention_21509196218952
// MI455X (gfx1250) — matrix operand order repaired and checked by reading, not compiled, not run on hardware
//
#include <hip/hip_runtime.h>
#include <hip/hip_bf16.h>

// ---------------------------------------------------------------------------
// MHA forward for MI455X (gfx1250), wave32 + WMMA bf16 + async LDS copies.
//   0) convert x, Wq..Wo to bf16 (one pass, packed)
//   1) q/k/v = x @ W{q,k,v}.T + b   (bf16 WMMA, f32 accum, double-buffered
//                                    async global->LDS tile staging)
//   2) attn  = flash-attention per (b, h) with online softmax
//   3) out   = attn @ Wo.T + bo     (f32 out to d_out)
// ---------------------------------------------------------------------------

typedef __bf16 bf16_t;
typedef __attribute__((ext_vector_type(16))) __bf16 v16bf;
typedef __attribute__((ext_vector_type(8)))  __bf16 v8bf;
typedef __attribute__((ext_vector_type(8)))  float  v8f;
typedef __attribute__((ext_vector_type(4)))  float  v4f;
typedef int v4i_g __attribute__((vector_size(16)));   // matches builtin param

#define BATCH  4
#define SEQ    2048
#define DMODEL 1024
#define NHEAD  16
#define HDIM   64

// ---- CDNA5 async global->LDS copy (ASYNCcnt DMA), guarded fallbacks -------
#if defined(__has_builtin)
#if __has_builtin(__builtin_amdgcn_global_load_async_to_lds_b128)
#define HAVE_ASYNC_COPY 1
#endif
#if __has_builtin(__builtin_amdgcn_s_wait_asynccnt)
#define HAVE_WAIT_ASYNC 1
#endif
#endif
#ifndef HAVE_ASYNC_COPY
#define HAVE_ASYNC_COPY 0
#endif
#ifndef HAVE_WAIT_ASYNC
#define HAVE_WAIT_ASYNC 0
#endif

#if HAVE_WAIT_ASYNC
#define WAIT_ASYNC(n) __builtin_amdgcn_s_wait_asynccnt(n)
#else
#define WAIT_ASYNC(n) asm volatile("s_wait_asynccnt %0" ::"n"(n) : "memory")
#endif

// one async 16-byte global->LDS DMA
static __device__ __forceinline__ void async_cp16B(const bf16_t* g, bf16_t* s) {
#if HAVE_ASYNC_COPY
  __builtin_amdgcn_global_load_async_to_lds_b128(
      (__attribute__((address_space(1))) v4i_g*)(g),
      (__attribute__((address_space(3))) v4i_g*)(s), 0, 0);
#else
  *(v8bf*)s = *(const v8bf*)g;
#endif
}

// copy 16 contiguous bf16 (32 B) global -> LDS
static __device__ __forceinline__ void stage16(const bf16_t* g, bf16_t* s) {
  async_cp16B(g, s);
  async_cp16B(g + 8, s + 8);
}

// xor-shuffle within 16-lane halves via ds_swizzle (group-of-32: xor|or|and)
template <int MASK>
static __device__ __forceinline__ float swz_xor(float v) {
  return __builtin_bit_cast(
      float, __builtin_amdgcn_ds_swizzle(__builtin_bit_cast(int, v),
                                         (MASK << 10) | 0x1f));
}

static __device__ __forceinline__ v16bf pack16(v8bf lo, v8bf hi) {
  v16bf f;
#pragma unroll
  for (int i = 0; i < 8; ++i) { f[i] = lo[i]; f[i + 8] = hi[i]; }
  return f;
}

// A-fragment (16x32 bf16, ISA 7.12.2): lane L holds row M=L%16,
// K = (L/16)*8 + {0..7} in VGPR0-3 and K = 16 + (L/16)*8 + {0..7} in VGPR4-7.
static __device__ __forceinline__ v16bf frag_ld_split(const bf16_t* p, int kb) {
  return pack16(*(const v8bf*)(p + kb), *(const v8bf*)(p + 16 + kb));
}

// contiguous-k fragment (not ISA table order; unused): lane L holds col N=L%16, K = (L/16)*16 + {0..15}
// contiguous -> 16 contiguous bf16 from an n-major tile row.
static __device__ __forceinline__ v16bf frag_ld_contig(const bf16_t* p) {
  return pack16(*(const v8bf*)p, *(const v8bf*)(p + 8));
}

// ---------------------------------------------------------------------------
// f32 -> bf16 one-pass convert, 8 elements/thread (n multiple of 8)
// ---------------------------------------------------------------------------
__global__ __launch_bounds__(256)
void cvt_f32_bf16(const float* __restrict__ in, bf16_t* __restrict__ out, int n) {
  const int i = (blockIdx.x * 256 + threadIdx.x) * 8;
  if (i < n) {
    v4f a = *(const v4f*)(in + i);
    v4f b = *(const v4f*)(in + i + 4);
    v8bf r;
#pragma unroll
    for (int j = 0; j < 4; ++j) { r[j] = (bf16_t)a[j]; r[j + 4] = (bf16_t)b[j]; }
    *(v8bf*)(out + i) = r;
  }
}

// ---------------------------------------------------------------------------
// GEMM: C[M,N] = A[M,K] @ W[N,K]^T + bias   (torch Linear; all bf16 operands)
// W[N,K] row-major is already the n-major B layout -> no transpose needed.
// Block tile 128x128x32, 256 threads = 8 waves (4x2), wave tile 32x64.
// Double-buffered LDS; staging via async global->LDS b128 DMA.
// ---------------------------------------------------------------------------
template <typename OT>
__global__ __launch_bounds__(256)
void gemm_xwT_wmma(const bf16_t* __restrict__ A,
                   const bf16_t* __restrict__ W,
                   const float* __restrict__ bias,
                   OT* __restrict__ C,
                   int M, int N, int K)
{
  constexpr int LP = 40;                 // LDS pitch in bf16 (80 B, 16B-mult)
  __shared__ bf16_t As[2][128 * LP];
  __shared__ bf16_t Bs[2][128 * LP];

  const int tid  = threadIdx.x;
  const int lane = tid & 31;
  const int wave = tid >> 5;
  const int wm   = wave >> 1;            // 0..3 (rows of 32)
  const int wn   = wave & 1;             // 0..1 (cols of 64)
  const int bm0  = blockIdx.y * 128;
  const int bn0  = blockIdx.x * 128;

  const int srow = tid >> 1;             // staging row 0..127
  const int scol = (tid & 1) * 16;       // staging col 0/16
  const int soff = srow * LP + scol;

  const int r  = lane & 15;
  const int kb = (lane >> 4) * 8;
  const int kh = (lane >> 4) * 16;

  const bf16_t* gA = A + (size_t)(bm0 + srow) * K + scol;
  const bf16_t* gW = W + (size_t)(bn0 + srow) * K + scol;

  v8f acc[2][4] = {};

  // prologue: stage tile 0 into buffer 0 (4 async ops per wave)
  stage16(gA, &As[0][soff]);
  stage16(gW, &Bs[0][soff]);

  int buf = 0;
  for (int k0 = 0; k0 < K; k0 += 32) {
    if (k0 + 32 < K) {                   // stage next tile into other buffer
      stage16(gA + k0 + 32, &As[buf ^ 1][soff]);
      stage16(gW + k0 + 32, &Bs[buf ^ 1][soff]);
      WAIT_ASYNC(4);                     // oldest 4 (current tile) complete
    } else {
      WAIT_ASYNC(0);
    }
    __syncthreads();

    v16bf af[2];
#pragma unroll
    for (int mt = 0; mt < 2; ++mt)
      af[mt] = frag_ld_split(&As[buf][(wm * 32 + mt * 16 + r) * LP], kb);

#pragma unroll
    for (int nt = 0; nt < 4; ++nt) {
      v16bf bfg = frag_ld_split(&Bs[buf][(wn * 64 + nt * 16 + r) * LP], kb);
#pragma unroll
      for (int mt = 0; mt < 2; ++mt)
        acc[mt][nt] = __builtin_amdgcn_wmma_f32_16x16x32_bf16(
            false, af[mt], false, bfg, (short)0, acc[mt][nt], false, false);
    }
    __syncthreads();                     // all waves done reading buf
    buf ^= 1;
  }

  // ---- epilogue: bias add; C layout lane(N=L%16), VGPR rr -> M=rr+(L/16)*8
#pragma unroll
  for (int mt = 0; mt < 2; ++mt) {
#pragma unroll
    for (int nt = 0; nt < 4; ++nt) {
      const int col  = bn0 + wn * 64 + nt * 16 + r;
      const float bv = bias[col];
      const int row0 = bm0 + wm * 32 + mt * 16 + (lane >> 4) * 8;
#pragma unroll
      for (int rr = 0; rr < 8; ++rr) {
        float v = acc[mt][nt][rr] + bv;
        C[(size_t)(row0 + rr) * N + col] = (OT)v;
      }
    }
  }
}

// ---------------------------------------------------------------------------
// Flash attention: grid (SEQ/128, NHEAD, BATCH), 256 threads = 8 waves.
// Each wave owns 16 q-rows; kv streamed in tiles of 64 with online softmax.
// K tile staged by async DMA (row-major = B-frag ready); V staged transposed.
// ---------------------------------------------------------------------------
__global__ __launch_bounds__(256)
void flash_attn_wmma(const bf16_t* __restrict__ Q,
                     const bf16_t* __restrict__ K,
                     const bf16_t* __restrict__ V,
                     bf16_t* __restrict__ O)
{
  __shared__ bf16_t Ks[64 * 64];         // [kv][d]   row-major
  __shared__ bf16_t Vst[64 * 64];        // [d][kv]   transposed
  __shared__ bf16_t Ps[8 * 16 * 64];     // per-wave P relayout slab

  const int tid  = threadIdx.x;
  const int lane = tid & 31;
  const int wave = tid >> 5;
  const int h    = blockIdx.y;
  const int b    = blockIdx.z;
  const int q0   = blockIdx.x * 128 + wave * 16;

  const int r    = lane & 15;
  const int half = lane >> 4;

  const size_t headoff = (size_t)h * HDIM;
  const bf16_t* Qb = Q + (size_t)b * SEQ * DMODEL + headoff;
  const bf16_t* Kb = K + (size_t)b * SEQ * DMODEL + headoff;
  const bf16_t* Vb = V + (size_t)b * SEQ * DMODEL + headoff;

  // ---- Q fragments: 2 chunks of d (0..31, 32..63), live in regs ----
  v16bf qf[2];
  {
    const bf16_t* qp = Qb + (size_t)(q0 + r) * DMODEL;
    const int qkb = half * 8;
#pragma unroll
    for (int c = 0; c < 2; ++c)
      qf[c] = frag_ld_split(qp + c * 32, qkb);
  }

  v8f o[4] = {};
  float mrow[8], lrow[8];
#pragma unroll
  for (int rr = 0; rr < 8; ++rr) { mrow[rr] = -1e30f; lrow[rr] = 0.f; }

  const float scale = 0.125f;            // 1/sqrt(64)

  const int svr = tid >> 2;              // staging kv row 0..63
  const int svc = (tid & 3) * 16;        // staging d col 0/16/32/48

  for (int kv0 = 0; kv0 < SEQ; kv0 += 64) {
    // ---- K tile via async DMA; V tile manual transpose into Vst ----
    stage16(Kb + (size_t)(kv0 + svr) * DMODEL + svc, Ks + svr * 64 + svc);
    {
      const bf16_t* vg = Vb + (size_t)(kv0 + svr) * DMODEL + svc;
      v8bf v0 = *(const v8bf*)vg;
      v8bf v1 = *(const v8bf*)(vg + 8);
#pragma unroll
      for (int i = 0; i < 8; ++i) {
        Vst[(svc + i)     * 64 + svr] = v0[i];
        Vst[(svc + 8 + i) * 64 + svr] = v1[i];
      }
    }
    WAIT_ASYNC(0);
    __syncthreads();

    // ---- S = (Q @ K^T) * scale ----
    v8f s[4] = {};
#pragma unroll
    for (int nt = 0; nt < 4; ++nt) {
#pragma unroll
      for (int c = 0; c < 2; ++c) {
        v16bf kf = frag_ld_split(Ks + (nt * 16 + r) * 64 + c * 32, half * 8);
        s[nt] = __builtin_amdgcn_wmma_f32_16x16x32_bf16(
            false, qf[c], false, kf, (short)0, s[nt], false, false);
      }
    }
#pragma unroll
    for (int nt = 0; nt < 4; ++nt) s[nt] = s[nt] * scale;

    // ---- online softmax; rows live in 16-lane halves -> ds_swizzle xor ----
    float alpha[8];
#pragma unroll
    for (int rr = 0; rr < 8; ++rr) {
      float mx = fmaxf(fmaxf(s[0][rr], s[1][rr]), fmaxf(s[2][rr], s[3][rr]));
      mx = fmaxf(mx, swz_xor<1>(mx));
      mx = fmaxf(mx, swz_xor<2>(mx));
      mx = fmaxf(mx, swz_xor<4>(mx));
      mx = fmaxf(mx, swz_xor<8>(mx));
      const float mnew = fmaxf(mrow[rr], mx);
      alpha[rr] = __expf(mrow[rr] - mnew);
      mrow[rr]  = mnew;
      float rs = 0.f;
#pragma unroll
      for (int nt = 0; nt < 4; ++nt) {
        s[nt][rr] = __expf(s[nt][rr] - mnew);
        rs += s[nt][rr];
      }
      rs += swz_xor<1>(rs);
      rs += swz_xor<2>(rs);
      rs += swz_xor<4>(rs);
      rs += swz_xor<8>(rs);
      lrow[rr] = lrow[rr] * alpha[rr] + rs;
    }
#pragma unroll
    for (int dt = 0; dt < 4; ++dt)
#pragma unroll
      for (int rr = 0; rr < 8; ++rr)
        o[dt][rr] *= alpha[rr];

    // ---- P (C-layout f32) -> bf16 A-fragments via wave-private LDS slab ----
    bf16_t* pw = Ps + wave * 16 * 64;
#pragma unroll
    for (int nt = 0; nt < 4; ++nt)
#pragma unroll
      for (int rr = 0; rr < 8; ++rr)
        pw[(half * 8 + rr) * 64 + nt * 16 + r] = (bf16_t)s[nt][rr];

    v16bf pf[2];
    {
      const bf16_t* pp = pw + r * 64;
      const int pkb = half * 8;
#pragma unroll
      for (int c = 0; c < 2; ++c)
        pf[c] = frag_ld_split(pp + c * 32, pkb);
    }

    // ---- O += P @ V ----
#pragma unroll
    for (int dt = 0; dt < 4; ++dt) {
#pragma unroll
      for (int c = 0; c < 2; ++c) {
        v16bf vf = frag_ld_split(Vst + (dt * 16 + r) * 64 + c * 32, half * 8);
        o[dt] = __builtin_amdgcn_wmma_f32_16x16x32_bf16(
            false, pf[c], false, vf, (short)0, o[dt], false, false);
      }
    }
    __syncthreads();
  }

  // ---- normalize, write head slice as bf16 ----
  bf16_t* ob = O + ((size_t)b * SEQ + q0) * DMODEL + headoff;
#pragma unroll
  for (int dt = 0; dt < 4; ++dt) {
    const int dcol = dt * 16 + r;
#pragma unroll
    for (int rr = 0; rr < 8; ++rr) {
      const int qr = half * 8 + rr;
      ob[(size_t)qr * DMODEL + dcol] = (bf16_t)(o[dt][rr] / lrow[rr]);
    }
  }
}

// ---------------------------------------------------------------------------
extern "C" void kernel_launch(void* const* d_in, const int* in_sizes, int n_in,
                              void* d_out, int out_size, void* d_ws, size_t ws_size,
                              hipStream_t stream) {
  (void)in_sizes; (void)n_in; (void)out_size; (void)ws_size;
  const float* x  = (const float*)d_in[0];
  const float* Wq = (const float*)d_in[1];
  const float* bq = (const float*)d_in[2];
  const float* Wk = (const float*)d_in[3];
  const float* bk = (const float*)d_in[4];
  const float* Wv = (const float*)d_in[5];
  const float* bv = (const float*)d_in[6];
  const float* Wo = (const float*)d_in[7];
  const float* bo = (const float*)d_in[8];
  float* out = (float*)d_out;

  const size_t elems = (size_t)BATCH * SEQ * DMODEL;   // 8,388,608
  const size_t welems = (size_t)DMODEL * DMODEL;       // 1,048,576
  bf16_t* Xb  = (bf16_t*)d_ws;
  bf16_t* Wqb = Xb  + elems;
  bf16_t* Wkb = Wqb + welems;
  bf16_t* Wvb = Wkb + welems;
  bf16_t* Wob = Wvb + welems;
  bf16_t* Qb  = Wob + welems;
  bf16_t* Kb  = Qb  + elems;
  bf16_t* Vb  = Kb  + elems;
  bf16_t* Ab  = Vb  + elems;                           // ~92 MB bf16 total

  dim3 blk(256);
  cvt_f32_bf16<<<dim3((int)(elems  / 2048)), blk, 0, stream>>>(x,  Xb,  (int)elems);
  cvt_f32_bf16<<<dim3((int)(welems / 2048)), blk, 0, stream>>>(Wq, Wqb, (int)welems);
  cvt_f32_bf16<<<dim3((int)(welems / 2048)), blk, 0, stream>>>(Wk, Wkb, (int)welems);
  cvt_f32_bf16<<<dim3((int)(welems / 2048)), blk, 0, stream>>>(Wv, Wvb, (int)welems);
  cvt_f32_bf16<<<dim3((int)(welems / 2048)), blk, 0, stream>>>(Wo, Wob, (int)welems);

  const int M = BATCH * SEQ, N = DMODEL, Kd = DMODEL;
  dim3 gg(N / 128, M / 128);

  gemm_xwT_wmma<bf16_t><<<gg, blk, 0, stream>>>(Xb, Wqb, bq, Qb, M, N, Kd);
  gemm_xwT_wmma<bf16_t><<<gg, blk, 0, stream>>>(Xb, Wkb, bk, Kb, M, N, Kd);
  gemm_xwT_wmma<bf16_t><<<gg, blk, 0, stream>>>(Xb, Wvb, bv, Vb, M, N, Kd);

  flash_attn_wmma<<<dim3(SEQ / 128, NHEAD, BATCH), blk, 0, stream>>>(Qb, Kb, Vb, Ab);

  gemm_xwT_wmma<float><<<gg, blk, 0, stream>>>(Ab, Wob, bo, out, M, N, Kd);
}
